// RouteNet_26431228739847
// MI455X (gfx1250) — matrix operand order repaired and checked by reading, not compiled, not run on hardware
//
#include <hip/hip_runtime.h>
#include <hip/hip_bf16.h>

typedef __attribute__((ext_vector_type(16))) _Float16 v16h;
typedef __attribute__((ext_vector_type(8)))  _Float16 v8h;
typedef __attribute__((ext_vector_type(8)))  float    v8f;

#define N_LINKS 10000
#define N_PATHS 100000
#define NEDGE   800000

// ---------------------------------------------------------------------------
// WMMA helper:  D = A(16x32 f16) * B(32x16 f16) + C(16x16 f32)
// ---------------------------------------------------------------------------
__device__ __forceinline__ v8f wmma16x16x32(v16h a, v16h b, v8f c) {
    return __builtin_amdgcn_wmma_f32_16x16x32_f16(
        /*neg_a=*/false, a, /*neg_b=*/false, b,
        /*c_mod=*/(short)0, c, /*reuse_a=*/false, /*reuse_b=*/false);
}

// A-matrix (16x32 f16) lane layout (ISA 7.12.2):
//   lanes 0-15 : row M=lane,    halves 0..7 -> K=0..7,   8..15 -> K=16..23
//   lanes 16-31: row M=lane-16, halves 0..7 -> K=8..15,  8..15 -> K=24..31
__device__ __forceinline__ v16h load_a_f32(const float* __restrict__ src,
                                           int row0, int lane) {
    const int r  = row0 + (lane & 15);
    const int kb = (lane >> 4) ? 8 : 0;
    const float4* p0 = (const float4*)(src + r * 32 + kb);       // K = kb..kb+7
    const float4* p1 = (const float4*)(src + r * 32 + kb + 16);  // K = kb+16..
    float4 a0 = p0[0], a1 = p0[1], b0 = p1[0], b1 = p1[1];
    v16h a;
    a[0]=(_Float16)a0.x;  a[1]=(_Float16)a0.y;  a[2]=(_Float16)a0.z;  a[3]=(_Float16)a0.w;
    a[4]=(_Float16)a1.x;  a[5]=(_Float16)a1.y;  a[6]=(_Float16)a1.z;  a[7]=(_Float16)a1.w;
    a[8]=(_Float16)b0.x;  a[9]=(_Float16)b0.y;  a[10]=(_Float16)b0.z; a[11]=(_Float16)b0.w;
    a[12]=(_Float16)b1.x; a[13]=(_Float16)b1.y; a[14]=(_Float16)b1.z; a[15]=(_Float16)b1.w;
    return a;
}

// Load A tile (K-chunk kc) from LDS f16 [16][256]: two 16-byte DS loads.
__device__ __forceinline__ v16h load_a_lds(const _Float16* __restrict__ src,
                                           int lane, int kc) {
    const int r  = lane & 15;
    const int kb = (lane >> 4) ? 8 : 0;
    v8h lo = *(const v8h*)(src + r * 256 + kc * 32 + kb);
    v8h hi = *(const v8h*)(src + r * 256 + kc * 32 + kb + 16);
    v16h a;
#pragma unroll
    for (int i = 0; i < 8; ++i) { a[i] = lo[i]; a[8 + i] = hi[i]; }
    return a;
}

// Pre-packed B tile: one contiguous 32-byte vector per lane.
__device__ __forceinline__ v16h load_b_packed(const _Float16* __restrict__ Wp,
                                              int tile_idx, int lane) {
    return ((const v16h*)Wp)[tile_idx * 32 + lane];
}

// ---------------------------------------------------------------------------
// Fast transcendentals: avoid the IEEE div-fixup chain, use HW rcp/tanh.
// ---------------------------------------------------------------------------
__device__ __forceinline__ float fast_rcp(float x) {
#if __has_builtin(__builtin_amdgcn_rcpf)
    return __builtin_amdgcn_rcpf(x);
#else
    return 1.0f / x;
#endif
}
__device__ __forceinline__ float sigmoidf_(float x) {
    return fast_rcp(1.0f + __expf(-x));
}
__device__ __forceinline__ float tanhf_(float x) {
#if __has_builtin(__builtin_amdgcn_tanhf)
    return __builtin_amdgcn_tanhf(x);
#else
    // tanh(x) = 1 - 2/(exp(2x)+1), numerically fine for GRU-range inputs
    return 1.0f - 2.0f * fast_rcp(__expf(2.0f * x) + 1.0f);
#endif
}
__device__ __forceinline__ float seluf_(float x) {
    const float scale = 1.0507009873554805f, alpha = 1.6732632423543772f;
    return scale * (x > 0.0f ? x : alpha * (__expf(x) - 1.0f));
}

// ---------------------------------------------------------------------------
// Pack W[K rows][ncols] (f32, row-major) into WMMA B-operand f16 layout:
//   Wp[((t*kchunks + kc)*32 + lane)*16 + i]
//     = W[(kc*32 + (lane>=16?8:0) + (i&7) + (i>=8?16:0))*ncols + t*16 + (lane&15)]
// ---------------------------------------------------------------------------
__global__ void pack_b_kernel(const float* __restrict__ W,
                              _Float16* __restrict__ Wp,
                              int ncols, int kchunks) {
    int idx = blockIdx.x * blockDim.x + threadIdx.x;
    int total = (ncols / 16) * kchunks * 32 * 16;
    if (idx >= total) return;
    int i     = idx & 15;
    int lane  = (idx >> 4) & 31;
    int chunk = idx >> 9;             // t*kchunks + kc
    int kc    = chunk % kchunks;
    int t     = chunk / kchunks;
    int krow  = kc * 32 + ((lane >> 4) ? 8 : 0) + (i & 7) + ((i >> 3) ? 16 : 0);
    Wp[idx] = (_Float16)W[krow * ncols + t * 16 + (lane & 15)];
}

// ---------------------------------------------------------------------------
// state init / zero
// ---------------------------------------------------------------------------
__global__ void init_state_kernel(const float* __restrict__ seed,
                                  float* __restrict__ state, int n) {
    int idx = blockIdx.x * blockDim.x + threadIdx.x;
    if (idx >= n * 32) return;
    state[idx] = ((idx & 31) == 0) ? seed[idx >> 5] : 0.0f;
}

__global__ void zero_kernel(float* __restrict__ buf, int n) {
    int idx = blockIdx.x * blockDim.x + threadIdx.x;
    if (idx < n) buf[idx] = 0.0f;
}

// ---------------------------------------------------------------------------
// segment-sum scatter: dst[didx[e]][d] += src[sidx[e]][d]   (32-wide rows)
// ---------------------------------------------------------------------------
__global__ void scatter_add_kernel(const float* __restrict__ src,
                                   float* __restrict__ dst,
                                   const int* __restrict__ sidx,
                                   const int* __restrict__ didx) {
    int idx = blockIdx.x * blockDim.x + threadIdx.x;
    if (idx >= NEDGE * 32) return;
    int e = idx >> 5, d = idx & 31;
    atomicAdd(&dst[didx[e] * 32 + d], src[sidx[e] * 32 + d]);
}

// ---------------------------------------------------------------------------
// Keras GRUCell (reset_after=True), 32 units, input dim 32, in-place h.
// One wave per 16-row tile; 12 WMMAs.
// ---------------------------------------------------------------------------
__global__ void gru_wmma_kernel(const float* __restrict__ x,
                                float* __restrict__ h,
                                const _Float16* __restrict__ Wp,
                                const _Float16* __restrict__ RWp,
                                const float* __restrict__ bias) // [2][96]
{
    const int lane  = threadIdx.x;
    const int row0  = blockIdx.x * 16;
    const int n     = lane & 15;
    const int mbase = (lane >> 4) ? 8 : 0;

    v16h ax = load_a_f32(x, row0, lane);
    v16h ah = load_a_f32(h, row0, lane);

#pragma unroll
    for (int s = 0; s < 2; ++s) {            // unit halves u = s*16..s*16+15
        v8f czero = {};
        v8f xz = wmma16x16x32(ax, load_b_packed(Wp,  s,     lane), czero);
        v8f hz = wmma16x16x32(ah, load_b_packed(RWp, s,     lane), czero);
        v8f xr = wmma16x16x32(ax, load_b_packed(Wp,  2 + s, lane), czero);
        v8f hr = wmma16x16x32(ah, load_b_packed(RWp, 2 + s, lane), czero);
        v8f xh = wmma16x16x32(ax, load_b_packed(Wp,  4 + s, lane), czero);
        v8f hh = wmma16x16x32(ah, load_b_packed(RWp, 4 + s, lane), czero);

        const int cz = s * 16, cr = 32 + s * 16, ch = 64 + s * 16;
        const float b0z = bias[cz + n], b1z = bias[96 + cz + n];
        const float b0r = bias[cr + n], b1r = bias[96 + cr + n];
        const float b0h = bias[ch + n], b1h = bias[96 + ch + n];
        const int u = s * 16 + n;

#pragma unroll
        for (int i = 0; i < 8; ++i) {
            const int row = row0 + mbase + i;
            float z = sigmoidf_(xz[i] + b0z + hz[i] + b1z);
            float r = sigmoidf_(xr[i] + b0r + hr[i] + b1r);
            float c = tanhf_(xh[i] + b0h + r * (hh[i] + b1h));
            float hold = h[row * 32 + u];
            h[row * 32 + u] = z * hold + (1.0f - z) * c;
        }
    }
}

// ---------------------------------------------------------------------------
// Fused readout:  h1 = selu(ps@W1+b1) ; h2 = selu(h1@W2+b2)
//                 out = [h2, ps] @ fin_w + fin_b
// One wave per 16-row tile.  16 + 128 WMMAs; h1 staged in LDS as f16.
// ---------------------------------------------------------------------------
__global__ void readout_wmma_kernel(const float* __restrict__ ps,
                                    const _Float16* __restrict__ w1p,
                                    const float* __restrict__ b1,
                                    const _Float16* __restrict__ w2p,
                                    const float* __restrict__ b2,
                                    const float* __restrict__ fw,   // [288][2]
                                    const float* __restrict__ fb,   // [2]
                                    float* __restrict__ out)        // [N][2]
{
    __shared__ _Float16 h1s[16 * 256];
    __shared__ float    part[2][16][16];

    const int lane  = threadIdx.x;
    const int row0  = blockIdx.x * 16;
    const int n     = lane & 15;
    const int mbase = (lane >> 4) ? 8 : 0;

    // ---- layer 1: [16x32] x [32x256] -> LDS (f16) ----
    v16h a1 = load_a_f32(ps, row0, lane);
#pragma unroll 1
    for (int t = 0; t < 16; ++t) {
        v8f czero = {};
        v8f d = wmma16x16x32(a1, load_b_packed(w1p, t, lane), czero);
        const float bb = b1[t * 16 + n];
#pragma unroll
        for (int i = 0; i < 8; ++i)
            h1s[(mbase + i) * 256 + t * 16 + n] = (_Float16)seluf_(d[i] + bb);
    }
    __syncthreads();

    // ---- layer 2 + final head partials ----
    float acc0[8], acc1[8];
#pragma unroll
    for (int i = 0; i < 8; ++i) { acc0[i] = 0.0f; acc1[i] = 0.0f; }

#pragma unroll 1
    for (int t2 = 0; t2 < 16; ++t2) {
        v8f acc = {};
#pragma unroll
        for (int kc = 0; kc < 8; ++kc) {
            v16h a2 = load_a_lds(h1s, lane, kc);
            v16h bt = load_b_packed(w2p, t2 * 8 + kc, lane);
            acc = wmma16x16x32(a2, bt, acc);
        }
        const float bb = b2[t2 * 16 + n];
        const float f0 = fw[(t2 * 16 + n) * 2 + 0];
        const float f1 = fw[(t2 * 16 + n) * 2 + 1];
#pragma unroll
        for (int i = 0; i < 8; ++i) {
            float v = seluf_(acc[i] + bb);
            acc0[i] += v * f0;
            acc1[i] += v * f1;
        }
    }
#pragma unroll
    for (int i = 0; i < 8; ++i) {
        part[0][mbase + i][n] = acc0[i];
        part[1][mbase + i][n] = acc1[i];
    }
    __syncthreads();

    // ---- cross-lane reduce + path_state tail of the head ----
    const int col = lane >> 4;       // output column 0 or 1
    const int row = lane & 15;
    float s = fb[col];
#pragma unroll
    for (int j = 0; j < 16; ++j) s += part[col][row][j];
    for (int p = 0; p < 32; ++p)
        s += ps[(row0 + row) * 32 + p] * fw[(256 + p) * 2 + col];
    out[(row0 + row) * 2 + col] = s;
}

// ---------------------------------------------------------------------------
extern "C" void kernel_launch(void* const* d_in, const int* in_sizes, int n_in,
                              void* d_out, int out_size, void* d_ws, size_t ws_size,
                              hipStream_t stream) {
    (void)in_sizes; (void)n_in; (void)out_size; (void)ws_size;

    const float* capacities   = (const float*)d_in[0];
    const float* traffic      = (const float*)d_in[1];
    const int*   links        = (const int*)d_in[2];
    const int*   paths        = (const int*)d_in[3];
    const float* path_kernel  = (const float*)d_in[6];
    const float* path_rkernel = (const float*)d_in[7];
    const float* path_bias    = (const float*)d_in[8];
    const float* link_kernel  = (const float*)d_in[9];
    const float* link_rkernel = (const float*)d_in[10];
    const float* link_bias    = (const float*)d_in[11];
    const float* ro_w1        = (const float*)d_in[12];
    const float* ro_b1        = (const float*)d_in[13];
    const float* ro_w2        = (const float*)d_in[14];
    const float* ro_b2        = (const float*)d_in[15];
    const float* fin_w        = (const float*)d_in[16];
    const float* fin_b        = (const float*)d_in[17];
    float* out = (float*)d_out;

    // ---- workspace layout ----
    float* path_state = (float*)d_ws;                    // N_PATHS*32 f32
    float* link_state = path_state + N_PATHS * 32;       // N_LINKS*32
    float* path_agg   = link_state + N_LINKS * 32;       // N_PATHS*32
    float* link_agg   = path_agg   + N_PATHS * 32;       // N_LINKS*32
    _Float16* fp      = (_Float16*)(link_agg + N_LINKS * 32);
    _Float16* pWp  = fp;               // 32*96  = 3072 halves
    _Float16* pRWp = pWp  + 3072;
    _Float16* lWp  = pRWp + 3072;
    _Float16* lRWp = lWp  + 3072;
    _Float16* w1p  = lRWp + 3072;      // 32*256 = 8192
    _Float16* w2p  = w1p  + 8192;      // 256*256 = 65536

    const int TPB = 256;

    // ---- one-time weight packing (f32 -> WMMA-layout f16) ----
    pack_b_kernel<<<(32 * 96  + TPB - 1) / TPB, TPB, 0, stream>>>(path_kernel,  pWp,  96,  1);
    pack_b_kernel<<<(32 * 96  + TPB - 1) / TPB, TPB, 0, stream>>>(path_rkernel, pRWp, 96,  1);
    pack_b_kernel<<<(32 * 96  + TPB - 1) / TPB, TPB, 0, stream>>>(link_kernel,  lWp,  96,  1);
    pack_b_kernel<<<(32 * 96  + TPB - 1) / TPB, TPB, 0, stream>>>(link_rkernel, lRWp, 96,  1);
    pack_b_kernel<<<(32 * 256 + TPB - 1) / TPB, TPB, 0, stream>>>(ro_w1,        w1p,  256, 1);
    pack_b_kernel<<<(256 * 256 + TPB - 1) / TPB, TPB, 0, stream>>>(ro_w2,       w2p,  256, 8);

    init_state_kernel<<<(N_PATHS * 32 + TPB - 1) / TPB, TPB, 0, stream>>>(
        traffic, path_state, N_PATHS);
    init_state_kernel<<<(N_LINKS * 32 + TPB - 1) / TPB, TPB, 0, stream>>>(
        capacities, link_state, N_LINKS);

    const int scat_blocks = (NEDGE * 32 + TPB - 1) / TPB;
    for (int t = 0; t < 8; ++t) {
        zero_kernel<<<(N_PATHS * 32 + TPB - 1) / TPB, TPB, 0, stream>>>(
            path_agg, N_PATHS * 32);
        scatter_add_kernel<<<scat_blocks, TPB, 0, stream>>>(
            link_state, path_agg, links, paths);
        gru_wmma_kernel<<<N_PATHS / 16, 32, 0, stream>>>(
            path_agg, path_state, pWp, pRWp, path_bias);

        zero_kernel<<<(N_LINKS * 32 + TPB - 1) / TPB, TPB, 0, stream>>>(
            link_agg, N_LINKS * 32);
        scatter_add_kernel<<<scat_blocks, TPB, 0, stream>>>(
            path_state, link_agg, paths, links);
        gru_wmma_kernel<<<N_LINKS / 16, 32, 0, stream>>>(
            link_agg, link_state, lWp, lRWp, link_bias);
    }

    readout_wmma_kernel<<<N_PATHS / 16, 32, 0, stream>>>(
        path_state, w1p, ro_b1, w2p, ro_b2, fin_w, fin_b, out);
}
